// base_encoder_38757784879438
// MI455X (gfx1250) — matrix operand order repaired and checked by reading, not compiled, not run on hardware
//
#include <hip/hip_runtime.h>
#include <hip/hip_bf16.h>

typedef __bf16 bf16;
typedef __attribute__((ext_vector_type(16))) __bf16 v16bf;
typedef __attribute__((ext_vector_type(8)))  float  v8f;

// ---------------------------------------------------------------------------
// Weight convert: Wt[out][in] (bf16) = W[in][out] (f32).  16384 elements.
// ---------------------------------------------------------------------------
__global__ void wt_to_bf16_t(const float* __restrict__ W, bf16* __restrict__ Wt) {
    int i = blockIdx.x * 256 + threadIdx.x;      // 0..16383
    int o = i >> 7, k = i & 127;
    Wt[i] = (bf16)W[k * 128 + o];
}

// c[i] = a[i] + b[i], 128 elements (combined per-dst-type conv biases)
__global__ void add_bias128(const float* __restrict__ a, const float* __restrict__ b,
                            float* __restrict__ c) {
    int i = threadIdx.x;
    c[i] = a[i] + b[i];
}

// ---------------------------------------------------------------------------
// Degree counting + reciprocal (mean = sum * (1/max(cnt,1)))
// ---------------------------------------------------------------------------
__global__ void count_deg(const int* __restrict__ dst, float* __restrict__ cnt, int n) {
    int i = blockIdx.x * 256 + threadIdx.x;
    if (i < n) atomicAdd(&cnt[dst[i]], 1.0f);
}

__global__ void recip_deg(float* __restrict__ c, int n) {
    int i = blockIdx.x * 256 + threadIdx.x;
    if (i < n) c[i] = 1.0f / fmaxf(c[i], 1.0f);
}

// ---------------------------------------------------------------------------
// Edge scatter-add: one wave (32 lanes) per edge, float4 per lane (128 feats)
// ---------------------------------------------------------------------------
__global__ __launch_bounds__(256) void scatter_add128(
    const float* __restrict__ x, const int* __restrict__ src,
    const int* __restrict__ dst, float* __restrict__ acc, int nedges) {
    long t = (long)blockIdx.x * 256 + threadIdx.x;
    long e = t >> 5;
    if (e >= nedges) return;
    int lane = (int)(t & 31);
    long sb = (long)src[e] * 128 + lane * 4;
    long db = (long)dst[e] * 128 + lane * 4;
    const float4 v = *reinterpret_cast<const float4*>(x + sb);
    atomicAdd(&acc[db + 0], v.x);
    atomicAdd(&acc[db + 1], v.y);
    atomicAdd(&acc[db + 2], v.z);
    atomicAdd(&acc[db + 3], v.w);
}

// ---------------------------------------------------------------------------
// Multi-source fused GEMM:  Y = act( sum_s (X_s * rowscale_s) @ W_s + bias ) + residual
// X_s: [nrows,128] f32; W_s: transposed bf16 [128out][128in]; all dims 128.
// Block: 256 threads (8 waves). Block = one 16-row tile; wave w = 16-col tile.
//
// A tile (16x128 f32 = 8KB) is staged into LDS with GLOBAL_LOAD_ASYNC_TO_LDS_B128
// (ASYNCcnt-tracked DMA copy). The per-lane LDS byte address is DERIVED FROM the
// Af pointer (ptrtoint low 32 bits == LDS offset), so Af is captured by the asm
// and the compiler must keep and order the subsequent ds_load reads.
// bf16 conversion + mean-scale happen during fragment build
// (ds_load pairs + v_cvt_pk_bf16_f32).
//
// Matrix core: v_wmma_f32_16x16x32_bf16, 4 K-steps per source, f32 accumulate.
// act: 1 = LeakyReLU(0.01), 0 = identity.
// ---------------------------------------------------------------------------
__global__ __launch_bounds__(256) void gemm128_multi(
    const float* __restrict__ X0, const bf16* __restrict__ W0, const float* __restrict__ s0,
    const float* __restrict__ X1, const bf16* __restrict__ W1, const float* __restrict__ s1,
    const float* __restrict__ X2, const bf16* __restrict__ W2, const float* __restrict__ s2,
    const float* __restrict__ X3, const bf16* __restrict__ W3, const float* __restrict__ s3,
    int nsrc, const float* __restrict__ bias, const float* __restrict__ residual,
    float* __restrict__ Y, int act) {
    __shared__ float Af[16 * 128];

    const int tid  = threadIdx.x;
    const int wave = tid >> 5;
    const int lane = tid & 31;
    const int half = lane >> 4;
    const int l16  = lane & 15;
    const long rowBase = (long)blockIdx.x * 16;

    // LDS byte address of Af, derived from the pointer itself (captures Af):
    // generic LDS-aperture address low 32 bits == LDS byte offset.
    const unsigned ldsBase = (unsigned)(size_t)(&Af[0]);
    const unsigned lds0 = ldsBase + (unsigned)(tid * 16);
    const unsigned lds1 = lds0 + 4096u;

    const float* Arow = Af + l16 * 128;

    v8f acc = {};

    for (int s = 0; s < nsrc; ++s) {
        const float* X  = (s == 0) ? X0 : (s == 1) ? X1 : (s == 2) ? X2 : X3;
        const bf16*  W  = (s == 0) ? W0 : (s == 1) ? W1 : (s == 2) ? W2 : W3;
        const float* sc = (s == 0) ? s0 : (s == 1) ? s1 : (s == 2) ? s2 : s3;

        // per-lane row scale for the mean normalization (hoisted, 1 load)
        float scr = 1.0f;
        if (sc) scr = sc[rowBase + l16];

        __syncthreads();   // previous source's readers done with LDS
        {
            // async DMA: 8KB tile = 256 lanes x 2 x 16B, global -> LDS
            const char* g0 = (const char*)(X + rowBase * 128) + tid * 16;
            const char* g1 = g0 + 4096;
            asm volatile(
                "global_load_async_to_lds_b128 %0, %1, off\n\t"
                "global_load_async_to_lds_b128 %2, %3, off"
                :: "v"(lds0), "v"(g0), "v"(lds1), "v"(g1)
                : "memory");
            asm volatile("s_wait_asynccnt 0x0" ::: "memory");
        }
        __syncthreads();

#pragma unroll
        for (int kk = 0; kk < 4; ++kk) {
            // A fragment: a[e] = A[row M = lane&15][K = half*16 + e] (contiguous K, matches B).
            v16bf a, b;
#pragma unroll
            for (int e = 0; e < 16; ++e) {
                int kA = kk * 32 + half * 16 + e;
                a[e] = (bf16)(Arow[kA] * scr);
            }
            // B fragment: b[e] = B[K = half*16 + e][col]; Wt is [col][K] -> contiguous
            const bf16* wp = W + ((long)(wave * 16 + l16)) * 128 + kk * 32 + half * 16;
#pragma unroll
            for (int e = 0; e < 16; ++e) b[e] = wp[e];

            acc = __builtin_amdgcn_wmma_f32_16x16x32_bf16(
                false, a, false, b, (short)0, acc, false, false);
        }
    }

    // Epilogue. C/D layout: VGPR r -> row M = half*8 + r, col N = lane&15
    const int n = wave * 16 + l16;
    const float bv = bias[n];
#pragma unroll
    for (int r = 0; r < 8; ++r) {
        long m = rowBase + half * 8 + r;
        float y = acc[r] + bv;
        if (act) y = (y >= 0.f) ? y : 0.01f * y;
        if (residual) y += residual[m * 128 + n];
        Y[m * 128 + n] = y;
    }
}

// ---------------------------------------------------------------------------
// Jumping-knowledge elementwise max
// ---------------------------------------------------------------------------
__global__ void eltmax(const float* __restrict__ a, const float* __restrict__ b,
                       float* __restrict__ c, long n) {
    long i = (long)blockIdx.x * 256 + threadIdx.x;
    if (i < n) c[i] = fmaxf(a[i], b[i]);
}

// ---------------------------------------------------------------------------
// Row L2 normalize (128 cols). One wave per row, 4 floats per lane.
// ---------------------------------------------------------------------------
__global__ __launch_bounds__(256) void l2norm_rows(const float* __restrict__ X,
                                                   float* __restrict__ Y, int nrows) {
    int row  = blockIdx.x * 8 + (threadIdx.x >> 5);
    int lane = threadIdx.x & 31;
    if (row >= nrows) return;
    const float* xr = X + (long)row * 128 + lane * 4;
    float x0 = xr[0], x1 = xr[1], x2 = xr[2], x3 = xr[3];
    float s = x0 * x0 + x1 * x1 + x2 * x2 + x3 * x3;
#pragma unroll
    for (int m = 16; m > 0; m >>= 1) s += __shfl_xor(s, m, 32);
    float inv = 1.0f / fmaxf(sqrtf(s), 1e-12f);
    float* yr = Y + (long)row * 128 + lane * 4;
    yr[0] = x0 * inv; yr[1] = x1 * inv; yr[2] = x2 * inv; yr[3] = x3 * inv;
}

// ---------------------------------------------------------------------------
// Host orchestration
// ---------------------------------------------------------------------------
extern "C" void kernel_launch(void* const* d_in, const int* in_sizes, int n_in,
                              void* d_out, int out_size, void* d_ws, size_t ws_size,
                              hipStream_t stream) {
    if (n_in < 34) return;

    const float* x_gene = (const float*)d_in[0];
    const float* x_dis  = (const float*)d_in[1];
    const int* gg_src = (const int*)d_in[2];
    const int* gg_dst = (const int*)d_in[3];
    const int* gd_src = (const int*)d_in[4];
    const int* gd_dst = (const int*)d_in[5];
    const int* dg_src = (const int*)d_in[6];
    const int* dg_dst = (const int*)d_in[7];
    const int Egg = in_sizes[2], Egd = in_sizes[4], Edg = in_sizes[6];
    const int NG  = in_sizes[0] / 128, ND = in_sizes[1] / 128;

    // params flattened in jax pytree order (dict keys sorted):
    // layers[0]: dg{Wl,Wr,b}=8,9,10  gd=11,12,13  gg=14,15,16
    // layers[1]: dg=17,18,19  gd=20,21,22  gg=23,24,25
    // post: disease{W,b}=26,27  gene=28,29 ; pre: disease=30,31 gene=32,33
    auto P = [&](int i) { return (const float*)d_in[i]; };

    // ---- workspace carving ----
    size_t off = 0;
    auto carve = [&](size_t bytes) -> void* {
        void* p = (char*)d_ws + off;
        off += (bytes + 255) & ~(size_t)255;
        return p;
    };
    // bf16 transposed weights: slots
    // 0:pre_g 1:pre_d | 2:l0ggWl 3:l0ggWr 4:l0dgWl 5:l0dgWr 6:l0gdWl 7:l0gdWr
    // 8..13: layer1 same | 14:post_g 15:post_d
    bf16* WtAll = (bf16*)carve((size_t)16 * 16384 * sizeof(bf16));
    auto Wt = [&](int s) { return WtAll + (size_t)s * 16384; };
    float* bias_g0 = (float*)carve(128 * 4);
    float* bias_g1 = (float*)carve(128 * 4);
    float* inv_gg  = (float*)carve((size_t)NG * 4);
    float* inv_dg  = (float*)carve((size_t)NG * 4);
    float* inv_gd  = (float*)carve((size_t)ND * 4);
    const size_t GB = (size_t)NG * 128 * 4, DB = (size_t)ND * 128 * 4;
    float* gA = (float*)carve(GB);   // pre out / JK result
    float* gBuf = (float*)carve(GB); // layer1 out
    float* gC = (float*)carve(GB);   // layer2 out
    float* dA = (float*)carve(DB);
    float* dBuf = (float*)carve(DB);
    float* dC = (float*)carve(DB);
    float* acc_gg = (float*)carve(GB);  // later reused as post-gene out
    float* acc_dg = (float*)carve(GB);
    float* acc_gd = (float*)carve(DB);  // later reused as post-disease out
    (void)ws_size; (void)out_size;

    // ---- convert weights (transposed bf16) ----
    const int wSrc[16] = {32, 30, 14, 15, 8, 9, 11, 12, 23, 24, 17, 18, 20, 21, 28, 26};
    for (int s = 0; s < 16; ++s)
        wt_to_bf16_t<<<64, 256, 0, stream>>>(P(wSrc[s]), Wt(s));

    // combined gene biases per layer: gg.b + dg.b
    add_bias128<<<1, 128, 0, stream>>>(P(16), P(10), bias_g0);
    add_bias128<<<1, 128, 0, stream>>>(P(25), P(19), bias_g1);

    // ---- degree counts -> reciprocals ----
    hipMemsetAsync(inv_gg, 0, (size_t)NG * 4, stream);
    hipMemsetAsync(inv_dg, 0, (size_t)NG * 4, stream);
    hipMemsetAsync(inv_gd, 0, (size_t)ND * 4, stream);
    count_deg<<<(Egg + 255) / 256, 256, 0, stream>>>(gg_dst, inv_gg, Egg);
    count_deg<<<(Edg + 255) / 256, 256, 0, stream>>>(dg_dst, inv_dg, Edg);
    count_deg<<<(Egd + 255) / 256, 256, 0, stream>>>(gd_dst, inv_gd, Egd);
    recip_deg<<<(NG + 255) / 256, 256, 0, stream>>>(inv_gg, NG);
    recip_deg<<<(NG + 255) / 256, 256, 0, stream>>>(inv_dg, NG);
    recip_deg<<<(ND + 255) / 256, 256, 0, stream>>>(inv_gd, ND);

    const int GT = NG / 16, DT = ND / 16;  // row tiles (NG, ND are multiples of 16)

    // ---- pre-MLP: leaky(X @ W + b) ----
    gemm128_multi<<<GT, 256, 0, stream>>>(
        x_gene, Wt(0), nullptr, nullptr, nullptr, nullptr, nullptr, nullptr, nullptr,
        nullptr, nullptr, nullptr, 1, P(33), nullptr, gA, 1);
    gemm128_multi<<<DT, 256, 0, stream>>>(
        x_dis, Wt(1), nullptr, nullptr, nullptr, nullptr, nullptr, nullptr, nullptr,
        nullptr, nullptr, nullptr, 1, P(31), nullptr, dA, 1);

    // ---- 2 SAGE layers ----
    const float* gin = gA; const float* din = dA;
    float* gout[2] = {gBuf, gC}; float* dout[2] = {dBuf, dC};
    for (int l = 0; l < 2; ++l) {
        int wl = 2 + l * 6;  // weight slot base for this layer
        hipMemsetAsync(acc_gg, 0, GB, stream);
        hipMemsetAsync(acc_dg, 0, GB, stream);
        hipMemsetAsync(acc_gd, 0, DB, stream);
        scatter_add128<<<(int)(((long)Egg * 32 + 255) / 256), 256, 0, stream>>>(
            gin, gg_src, gg_dst, acc_gg, Egg);
        scatter_add128<<<(int)(((long)Edg * 32 + 255) / 256), 256, 0, stream>>>(
            din, dg_src, dg_dst, acc_dg, Edg);
        scatter_add128<<<(int)(((long)Egd * 32 + 255) / 256), 256, 0, stream>>>(
            gin, gd_src, gd_dst, acc_gd, Egd);

        // gene dst: leaky(mean_gg@Wl_gg + g@Wr_gg + mean_dg@Wl_dg + g@Wr_dg + b) + g
        gemm128_multi<<<GT, 256, 0, stream>>>(
            acc_gg, Wt(wl + 0), inv_gg,
            gin,    Wt(wl + 1), nullptr,
            acc_dg, Wt(wl + 2), inv_dg,
            gin,    Wt(wl + 3), nullptr,
            4, (l == 0) ? bias_g0 : bias_g1, gin, gout[l], 1);
        // disease dst: leaky(mean_gd@Wl_gd + d@Wr_gd + b) + d
        gemm128_multi<<<DT, 256, 0, stream>>>(
            acc_gd, Wt(wl + 4), inv_gd,
            din,    Wt(wl + 5), nullptr,
            nullptr, nullptr, nullptr, nullptr, nullptr, nullptr,
            2, P(l == 0 ? 13 : 22), din, dout[l], 1);
        gin = gout[l]; din = dout[l];
    }

    // ---- jumping knowledge: elementwise max of the two layer outputs ----
    eltmax<<<(int)(((long)NG * 128 + 255) / 256), 256, 0, stream>>>(gBuf, gC, gA, (long)NG * 128);
    eltmax<<<(int)(((long)ND * 128 + 255) / 256), 256, 0, stream>>>(dBuf, dC, dA, (long)ND * 128);

    // ---- post-MLP (plain linear), reuse acc buffers for outputs ----
    gemm128_multi<<<GT, 256, 0, stream>>>(
        gA, Wt(14), nullptr, nullptr, nullptr, nullptr, nullptr, nullptr, nullptr,
        nullptr, nullptr, nullptr, 1, P(29), nullptr, acc_gg, 0);
    gemm128_multi<<<DT, 256, 0, stream>>>(
        dA, Wt(15), nullptr, nullptr, nullptr, nullptr, nullptr, nullptr, nullptr,
        nullptr, nullptr, nullptr, 1, P(27), nullptr, acc_gd, 0);

    // ---- L2 normalize into d_out: [gene block | disease block] ----
    float* outp = (float*)d_out;
    l2norm_rows<<<(NG + 7) / 8, 256, 0, stream>>>(acc_gg, outp, NG);
    l2norm_rows<<<(ND + 7) / 8, 256, 0, stream>>>(acc_gd, outp + (size_t)NG * 128, ND);
}
